// SingleHeadAttention_37349035606567
// MI455X (gfx1250) — matrix operand order repaired and checked by reading, not compiled, not run on hardware
//
#include <hip/hip_runtime.h>

#define BATCH 8
#define TSEQ  2048
#define CDIM  1024
#define HDIM  64
#define BT    (BATCH * TSEQ)

typedef __attribute__((ext_vector_type(16))) __bf16          v16bf;
typedef __attribute__((ext_vector_type(8)))  float           v8f;
typedef __attribute__((ext_vector_type(8)))  unsigned short  v8us;

union FragBF { v16bf v; unsigned short u[16]; unsigned int d[8]; v8us h[2]; };
union FragF  { v8f   v; float f[8]; };
union ARaw   { struct { v8f lo, hi; } v; float f[16]; };

__device__ __forceinline__ unsigned short f2bf(float f) {
    unsigned int u = __builtin_bit_cast(unsigned int, f);
    u += 0x7FFFu + ((u >> 16) & 1u);          // round-to-nearest-even
    return (unsigned short)(u >> 16);
}

// Packed f32x2 -> bf16x2 in one dword: round-to-nearest via +0x8000 (carry
// ripples into exponent correctly), then one v_perm_b32 grabs both high
// halves.  3 VALU ops per pair.
__device__ __forceinline__ unsigned int f2bf_pk(float lo, float hi) {
#if __has_builtin(__builtin_amdgcn_perm)
    const unsigned int ulo = __builtin_bit_cast(unsigned int, lo) + 0x8000u;
    const unsigned int uhi = __builtin_bit_cast(unsigned int, hi) + 0x8000u;
    // D = {S0[31:16], S1[31:16]} : low16 = lo_hi16, high16 = hi_hi16.
    return __builtin_amdgcn_perm(uhi, ulo, 0x07060302u);
#else
    return (unsigned int)f2bf(lo) | ((unsigned int)f2bf(hi) << 16);
#endif
}

__device__ __forceinline__ void load_bfrag(FragBF& b, const unsigned short* p) {
    const v8us* pv = (const v8us*)p;
    b.h[0] = pv[0];
    b.h[1] = pv[1];
}

// ---------------------------------------------------------------------------
// Kernel 0: one-time weight prep.  W*:[C,H] f32 -> wt bf16 [3][H][C] so
// projection B-fragments become contiguous 32B global loads.  Each thread
// produces one packed dword (two consecutive k for one n).
// ---------------------------------------------------------------------------
__global__ __launch_bounds__(256)
void wprep_kernel(const float* __restrict__ Wq,
                  const float* __restrict__ Wk,
                  const float* __restrict__ Wv,
                  unsigned int* __restrict__ wt32) {
    const int i = blockIdx.x * 256 + threadIdx.x;     // [0, 3*64*512) pairs
    if (i >= 3 * HDIM * (CDIM / 2)) return;
    const int w   = i / (HDIM * (CDIM / 2));
    const int rem = i % (HDIM * (CDIM / 2));
    const int n   = rem / (CDIM / 2);
    const int k2  = (rem % (CDIM / 2)) * 2;
    const float* wp = (w == 0) ? Wq : ((w == 1) ? Wk : Wv);
    wt32[i] = f2bf_pk(wp[(size_t)k2 * HDIM + n], wp[(size_t)(k2 + 1) * HDIM + n]);
}

// ---------------------------------------------------------------------------
// Kernel 1: Q/K/V projections, software-pipelined.
// kc loop manually unrolled x2 with two NAMED ARaw buffers (no dynamic
// register-array indexing -> no scratch).  B fragments double-buffered with
// cross-chunk wraparound.  A conversion: 3 VALU per element pair.
// ---------------------------------------------------------------------------
__global__ __launch_bounds__(256)
void qkv_proj_kernel(const float* __restrict__ x,
                     const unsigned short* __restrict__ wt,
                     unsigned short* __restrict__ qout,
                     unsigned short* __restrict__ kout,
                     unsigned short* __restrict__ vtout) {
    __shared__ __align__(16) unsigned short vstage[HDIM][128];  // 16 KB
    const int tid     = threadIdx.x;
    const int lane    = tid & 31;
    const int wave    = tid >> 5;
    const int lrow    = lane & 15;
    const int lhi     = lane >> 4;                 // 0 or 1
    const int rowBase = blockIdx.x * 128 + wave * 16;

    FragF acc[3][4];
    #pragma unroll
    for (int w = 0; w < 3; ++w)
        #pragma unroll
        for (int nt = 0; nt < 4; ++nt)
            #pragma unroll
            for (int i = 0; i < 8; ++i) acc[w][nt].f[i] = 0.0f;

    const float* xrow = x + (size_t)(rowBase + lrow) * CDIM + lhi * 16;
    // Base of this lane's B-fragment run: wt[w][nt*16+lrow][kc+lhi*16].
    const unsigned short* wbase = wt + (size_t)lrow * CDIM + lhi * 16;

    FragBF b[2];

    auto loadA = [&](ARaw& arr, int kc) {
        arr.v.lo = *(const v8f*)(xrow + kc);
        arr.v.hi = *(const v8f*)(xrow + kc + 8);
    };
    auto do_chunk = [&](const ARaw& arc, int kc, bool wrapNextB) {
        FragBF a;
        #pragma unroll
        for (int j = 0; j < 4; ++j) {
            a.d[j]     = f2bf_pk(arc.f[2 * j],     arc.f[2 * j + 1]);
            a.d[4 + j] = f2bf_pk(arc.f[8 + 2 * j], arc.f[8 + 2 * j + 1]);
        }
        #pragma unroll
        for (int idx = 0; idx < 12; ++idx) {
            if (idx + 1 < 12) {
                const int w2 = (idx + 1) >> 2, nt2 = (idx + 1) & 3;
                load_bfrag(b[(idx + 1) & 1],
                           wbase + (size_t)(w2 * HDIM + nt2 * 16) * CDIM + kc);
            } else if (wrapNextB) {
                load_bfrag(b[0], wbase + kc + 32);  // idx 0 of next chunk
            }
            acc[idx >> 2][idx & 3].v = __builtin_amdgcn_wmma_f32_16x16x32_bf16(
                false, a.v, false, b[idx & 1].v, (short)0,
                acc[idx >> 2][idx & 3].v, false, false);
        }
    };

    ARaw arA, arB;
    loadA(arA, 0);
    load_bfrag(b[0], wbase);                       // idx 0, kc 0
    for (int kc = 0; kc < CDIM; kc += 64) {
        loadA(arB, kc + 32);                       // prefetch odd chunk
        do_chunk(arA, kc, true);
        if (kc + 64 < CDIM) loadA(arA, kc + 64);   // prefetch next even chunk
        do_chunk(arB, kc + 32, kc + 64 < CDIM);
    }

    // Q (pre-scaled by C^-0.5) and K: row-major bf16, coalesced across lanes.
    #pragma unroll
    for (int nt = 0; nt < 4; ++nt)
        #pragma unroll
        for (int r = 0; r < 8; ++r) {
            const int m    = lhi * 8 + r;
            const size_t o = (size_t)(rowBase + m) * HDIM + nt * 16 + lrow;
            qout[o] = f2bf(acc[0][nt].f[r] * 0.03125f);
            kout[o] = f2bf(acc[1][nt].f[r]);
        }

    // V: transpose through LDS (packed dword stores), then coalesced global
    // stores to [B][H][T].
    #pragma unroll
    for (int nt = 0; nt < 4; ++nt)
        #pragma unroll
        for (int j = 0; j < 4; ++j)
            *(unsigned int*)&vstage[nt * 16 + lrow][wave * 16 + lhi * 8 + 2 * j] =
                f2bf_pk(acc[2][nt].f[2 * j], acc[2][nt].f[2 * j + 1]);
    __syncthreads();
    {
        const int bb = (blockIdx.x * 128) / TSEQ;
        const int t0 = (blockIdx.x * 128) & (TSEQ - 1);
        for (int i = tid; i < HDIM * 128 / 8; i += 256) {   // 16B chunks
            const int h   = i >> 4;
            const int seg = i & 15;
            *(v8us*)(vtout + ((size_t)bb * HDIM + h) * TSEQ + t0 + seg * 8) =
                *(const v8us*)&vstage[h][seg * 8];
        }
    }
}

// ---------------------------------------------------------------------------
// Kernel 2: causal flash attention.  One wave per 16 query rows.
// All K/V fragments for a key block issued up front; row-sum computed by an
// extra WMMA against an all-ones B fragment (online, scaled like O).
// ---------------------------------------------------------------------------
__global__ __launch_bounds__(32)
void attn_kernel(const unsigned short* __restrict__ Q,
                 const unsigned short* __restrict__ K,
                 const unsigned short* __restrict__ Vt,
                 float* __restrict__ out) {
    __shared__ __align__(16) unsigned short Plds[16][32];  // [query m][key ss]
    const int lane = threadIdx.x & 31;
    const int lrow = lane & 15;
    const int lhi  = lane >> 4;
    const int b    = blockIdx.y;
    const int q0   = blockIdx.x * 16;

    const unsigned short* Qb  = Q  + (size_t)b * TSEQ * HDIM;
    const unsigned short* Kb  = K  + (size_t)b * TSEQ * HDIM;
    const unsigned short* Vtb = Vt + (size_t)b * HDIM * TSEQ;  // [h][t]

    // Q A-fragments for h 0..31 and 32..63 (row = q0+lrow).
    FragBF aQ0, aQ1;
    {
        const v8us* p = (const v8us*)(Qb + (size_t)(q0 + lrow) * HDIM);
        aQ0.h[0] = p[2 * lhi];
        aQ0.h[1] = p[2 * lhi + 1];
        aQ1.h[0] = p[4 + 2 * lhi];
        aQ1.h[1] = p[5 + 2 * lhi];
    }
    FragBF ones;                                   // all-ones B frag (bf16 1.0)
    #pragma unroll
    for (int j = 0; j < 16; ++j) ones.u[j] = 0x3F80u;

    FragF o[4], lacc;
    #pragma unroll
    for (int ht = 0; ht < 4; ++ht)
        #pragma unroll
        for (int i = 0; i < 8; ++i) o[ht].f[i] = 0.0f;
    #pragma unroll
    for (int i = 0; i < 8; ++i) lacc.f[i] = 0.0f;
    float mrow[8];
    #pragma unroll
    for (int r = 0; r < 8; ++r) mrow[r] = -1e30f;

    for (int sb = 0; sb <= q0 + 15; sb += 32) {
        // Issue all fragment loads for this key block up front.
        FragBF bk[4], bv[4];
        #pragma unroll
        for (int j = 0; j < 2; ++j) {
            const unsigned short* kr =
                Kb + (size_t)(sb + j * 16 + lrow) * HDIM + lhi * 16;
            load_bfrag(bk[2 * j],     kr);
            load_bfrag(bk[2 * j + 1], kr + 32);
        }
        #pragma unroll
        for (int ht = 0; ht < 4; ++ht)
            load_bfrag(bv[ht], Vtb + (size_t)(ht * 16 + lrow) * TSEQ
                                   + sb + lhi * 16);
        // Prefetch next key block into caches.
        if (sb + 32 <= q0 + 15) {
            __builtin_prefetch(Kb + (size_t)(sb + 32 + lane) * HDIM, 0, 3);
            __builtin_prefetch(Vtb + (size_t)lane        * TSEQ + sb + 32, 0, 3);
            __builtin_prefetch(Vtb + (size_t)(32 + lane) * TSEQ + sb + 32, 0, 3);
        }

        // S tiles: S = (Q*scale)(16x64) . K_tile^T.
        FragF s[2];
        #pragma unroll
        for (int j = 0; j < 2; ++j) {
            FragF z;
            #pragma unroll
            for (int i = 0; i < 8; ++i) z.f[i] = 0.0f;
            v8f sv = __builtin_amdgcn_wmma_f32_16x16x32_bf16(
                false, aQ0.v, false, bk[2 * j].v, (short)0, z.v, false, false);
            sv = __builtin_amdgcn_wmma_f32_16x16x32_bf16(
                false, aQ1.v, false, bk[2 * j + 1].v, (short)0, sv, false, false);
            s[j].v = sv;
        }

        // Causal mask (only near the diagonal) + online softmax.
        const bool needMask = (sb + 31 > q0);      // uniform over the wave
        #pragma unroll
        for (int r = 0; r < 8; ++r) {
            float s0 = s[0].f[r];
            float s1 = s[1].f[r];
            if (needMask) {
                const int q = q0 + lhi * 8 + r;
                if (sb + lrow      > q) s0 = -1e30f;
                if (sb + 16 + lrow > q) s1 = -1e30f;
            }
            float mloc = fmaxf(s0, s1);
            #pragma unroll
            for (int d = 1; d < 16; d <<= 1)
                mloc = fmaxf(mloc, __shfl_xor(mloc, d, 32));
            const float mnew  = fmaxf(mrow[r], mloc);
            const float alpha = __expf(mrow[r] - mnew);
            mrow[r] = mnew;
            const float p0 = __expf(s0 - mnew);
            const float p1 = __expf(s1 - mnew);
            lacc.f[r] *= alpha;
            #pragma unroll
            for (int ht = 0; ht < 4; ++ht) o[ht].f[r] *= alpha;
            const int m = lhi * 8 + r;
            Plds[m][lrow]      = f2bf(p0);
            Plds[m][16 + lrow] = f2bf(p1);
        }

        // P as A-fragment (16x32); in-wave LDS ordering makes stores visible.
        FragBF aP;
        {
            const v8us* p0 = (const v8us*)&Plds[lrow][lhi * 16];
            const v8us* p1 = (const v8us*)&Plds[lrow][lhi * 16 + 8];
            aP.h[0] = p0[0];
            aP.h[1] = p1[0];
        }
        // O += P*V  and  l += P*1 (row-sum via WMMA instead of shuffles).
        #pragma unroll
        for (int ht = 0; ht < 4; ++ht)
            o[ht].v = __builtin_amdgcn_wmma_f32_16x16x32_bf16(
                false, aP.v, false, bv[ht].v, (short)0, o[ht].v, false, false);
        lacc.v = __builtin_amdgcn_wmma_f32_16x16x32_bf16(
            false, aP.v, false, ones.v, (short)0, lacc.v, false, false);
    }

    // Normalize and store f32 output [B][T][H].
    float* orow = out + ((size_t)b * TSEQ + q0) * HDIM;
    #pragma unroll
    for (int ht = 0; ht < 4; ++ht)
        #pragma unroll
        for (int r = 0; r < 8; ++r) {
            const int m = lhi * 8 + r;
            orow[(size_t)m * HDIM + ht * 16 + lrow] = o[ht].f[r] / lacc.f[r];
        }
}

// ---------------------------------------------------------------------------
extern "C" void kernel_launch(void* const* d_in, const int* in_sizes, int n_in,
                              void* d_out, int out_size, void* d_ws, size_t ws_size,
                              hipStream_t stream) {
    const float* x  = (const float*)d_in[0];
    const float* Wq = (const float*)d_in[1];
    const float* Wk = (const float*)d_in[2];
    const float* Wv = (const float*)d_in[3];

    unsigned short* qws = (unsigned short*)d_ws;            // 2 MB  [B][T][H]
    unsigned short* kws = qws + (size_t)BT * HDIM;          // 2 MB  [B][T][H]
    unsigned short* vws = kws + (size_t)BT * HDIM;          // 2 MB  [B][H][T]
    unsigned short* wtw = vws + (size_t)BT * HDIM;          // 384KB [3][H][C]

    wprep_kernel<<<dim3((3 * HDIM * (CDIM / 2) + 255) / 256), dim3(256), 0,
                   stream>>>(Wq, Wk, Wv, (unsigned int*)wtw);
    qkv_proj_kernel<<<dim3(BT / 128), dim3(256), 0, stream>>>(
        x, wtw, qws, kws, vws);
    attn_kernel<<<dim3(TSEQ / 16, BATCH), dim3(32), 0, stream>>>(
        qws, kws, vws, (float*)d_out);
}
